// Qwen3_5MoeBlock_77300821393706
// MI455X (gfx1250) — matrix operand order repaired and checked by reading, not compiled, not run on hardware
//
#include <hip/hip_runtime.h>
#include <hip/hip_bf16.h>
#include <stdint.h>

// ---------------------------------------------------------------------------
// Qwen3.5 MoE block for MI455X (gfx1250, wave32, WMMA + async-to-LDS staging).
// Memory-bound on fp32 weight streaming (~690 MB @ 23.3 TB/s ~= 30us): convert
// to bf16 on the fly, drive v_wmma_f32_16x16x32_bf16, stage weight panels with
// global_load_async_to_lds_b128 (ASYNCcnt), write d_out exactly once.
// Workspace (~250 MB): comb[N,E] f32 | sg[N] f32 | hb[N,H] bf16 |
//                      act[E,N,I_MOE] bf16 | acts[N,I_SH] bf16
// ---------------------------------------------------------------------------

typedef __attribute__((ext_vector_type(16))) __bf16 v16bf;
typedef __attribute__((ext_vector_type(8)))  __bf16 v8bf;
typedef __attribute__((ext_vector_type(8)))  float  v8f;

#define WMMA_BF16(a, b, c) \
  __builtin_amdgcn_wmma_f32_16x16x32_bf16(false, (a), false, (b), (short)0, (c), false, false)

static constexpr int N_TOK = 2 * 2048;   // B*S
static constexpr int HDIM  = 2048;
static constexpr int NEXP  = 16;
static constexpr int TOPK  = 4;
static constexpr int I_MOE = 1408;
static constexpr int I_SH  = 5632;

// ---------------------------------------------------------------------------
// gfx1250 async copy: per-lane 16B global->LDS, tracked by ASYNCcnt.
// VGLOBAL async encoding: vdst = LDS byte address VGPR, vaddr = 64b address.
// (flat->LDS mapping uses addr[31:0] as the LDS byte offset, ISA §10.2)
// ---------------------------------------------------------------------------
__device__ __forceinline__ void async_b128(unsigned lds_byte_addr, const void* src) {
  asm volatile("global_load_async_to_lds_b128 %0, %1, off"
               :: "v"(lds_byte_addr), "v"(src) : "memory");
}
__device__ __forceinline__ void wait_async_all() {
  asm volatile("s_wait_asynccnt 0" ::: "memory");
}

// Cooperatively stage a 32-row x 64-col fp32 panel of row-major W (leading
// dim ld) into LDS. 128 threads: 16 lanes cover one contiguous 256B row.
__device__ __forceinline__ void stage_panel(float* smem /* [32*64] */,
                                            const float* __restrict__ W, int ld) {
  const int t   = threadIdx.x;
  const int seg = t & 15;                       // which 16B chunk of the row
  const int r0  = t >> 4;                       // 0..7
  const unsigned lds_base = (unsigned)(uintptr_t)smem;   // low 32b = LDS offset
#pragma unroll
  for (int i = 0; i < 4; ++i) {
    const int row = r0 + i * 8;
    async_b128(lds_base + (unsigned)(row * 64 + seg * 4) * 4u,
               (const void*)(W + (size_t)row * ld + seg * 4));
  }
}

// ---------------------------------------------------------------------------
// WMMA fragment builders (layouts per CDNA5 ISA §7.12.2, wave32)
// ---------------------------------------------------------------------------

// A 16x32 bf16 fragment from row-major bf16 memory: two contiguous b128 loads.
__device__ __forceinline__ v16bf load_a_frag(const __bf16* __restrict__ tile, int lda) {
  const int lane = threadIdx.x & 31;
  const int row  = lane & 15;
  const int kh   = (lane >> 4) << 3;            // 0 or 8
  const __bf16* p = tile + (size_t)row * lda + kh;
  v8bf lo = *(const v8bf*)(p);                  // K = kh .. kh+7
  v8bf hi = *(const v8bf*)(p + 16);             // K = 16+kh .. 16+kh+7
  return __builtin_shufflevector(lo, hi, 0, 1, 2, 3, 4, 5, 6, 7,
                                         8, 9, 10, 11, 12, 13, 14, 15);
}

// B 32x16 fragment from a staged 32x64 fp32 LDS panel (wave picks its 16 cols).
__device__ __forceinline__ v16bf frag_b_from_lds(const float* __restrict__ smem, int wave) {
  const int lane = threadIdx.x & 31;
  const int col  = wave * 16 + (lane & 15);
  const int kb   = (lane >> 4) << 3;            // 0 or 8
  v16bf b;
#pragma unroll
  for (int j = 0; j < 16; ++j) b[j] = (__bf16)smem[(kb + j + (j >= 8 ? 8 : 0)) * 64 + col];
  return b;
}

// ---------------------------------------------------------------------------
__global__ void cvt_bf16_kernel(const float* __restrict__ x, __bf16* __restrict__ y, size_t n) {
  size_t i = (size_t)blockIdx.x * blockDim.x + threadIdx.x;
  size_t s = (size_t)gridDim.x * blockDim.x;
  for (; i < n; i += s) y[i] = (__bf16)x[i];
}

// ---------------------------------------------------------------------------
// Router: one wave32 per token. fp32 softmax over 16 logits, top-4, renorm.
// Also sg[n] = sigmoid(h . sh_gate_w).
// ---------------------------------------------------------------------------
__global__ __launch_bounds__(256)
void router_kernel(const float* __restrict__ h,      // [N,H]
                   const float* __restrict__ rw,     // [E,H]
                   const float* __restrict__ sgw,    // [H]
                   float* __restrict__ comb,         // [N,E]
                   float* __restrict__ sg) {         // [N]
  const int tok  = blockIdx.x * (blockDim.x >> 5) + (threadIdx.x >> 5);
  const int lane = threadIdx.x & 31;
  if (tok >= N_TOK) return;
  const float* hp = h + (size_t)tok * HDIM;

  float acc[NEXP];
#pragma unroll
  for (int e = 0; e < NEXP; ++e) acc[e] = 0.0f;
  float accg = 0.0f;

  for (int k = lane; k < HDIM; k += 32) {
    const float hv = hp[k];
#pragma unroll
    for (int e = 0; e < NEXP; ++e) acc[e] += hv * rw[(size_t)e * HDIM + k];
    accg += hv * sgw[k];
  }
#pragma unroll
  for (int off = 16; off > 0; off >>= 1) {      // wave32 butterfly
#pragma unroll
    for (int e = 0; e < NEXP; ++e) acc[e] += __shfl_xor(acc[e], off, 32);
    accg += __shfl_xor(accg, off, 32);
  }

  if (lane == 0) {
    float m = acc[0];
#pragma unroll
    for (int e = 1; e < NEXP; ++e) m = fmaxf(m, acc[e]);
    float p[NEXP], sum = 0.0f;
#pragma unroll
    for (int e = 0; e < NEXP; ++e) { p[e] = __expf(acc[e] - m); sum += p[e]; }
    const float inv = 1.0f / sum;
#pragma unroll
    for (int e = 0; e < NEXP; ++e) p[e] *= inv;

    float outw[NEXP];
#pragma unroll
    for (int e = 0; e < NEXP; ++e) outw[e] = 0.0f;
    float tsum = 0.0f;
    for (int t = 0; t < TOPK; ++t) {            // top-4 by repeated argmax
      int bi = 0; float bv = -1.0f;
#pragma unroll
      for (int e = 0; e < NEXP; ++e)
        if (p[e] > bv) { bv = p[e]; bi = e; }
      outw[bi] = bv; p[bi] = -2.0f; tsum += bv;
    }
    const float rinv = 1.0f / fmaxf(tsum, 1e-9f);
#pragma unroll
    for (int e = 0; e < NEXP; ++e) comb[(size_t)tok * NEXP + e] = outw[e] * rinv;
    sg[tok] = 1.0f / (1.0f + __expf(-accg));
  }
}

// ---------------------------------------------------------------------------
// Fused gate/up + SiLU: act[e,n,i] = bf16(silu(h@wg) * (h@wu)).
// Block = 128 (4 waves) owns a 64x64 tile; per 32-deep K slice the block
// async-stages both weight panels to LDS, then each wave runs 8 WMMA
// (4 A-fragments x {gate,up}).  grid = (I/64, N/64, E).
// ---------------------------------------------------------------------------
__global__ __launch_bounds__(128)
void moe_gateup_kernel(const __bf16* __restrict__ hb,   // [N,H] bf16
                       const float*  __restrict__ wg,   // [E,H,I] f32
                       const float*  __restrict__ wu,   // [E,H,I] f32
                       __bf16* __restrict__ act,        // [E,N,I] bf16
                       int I) {
  __shared__ float sGate[32 * 64];
  __shared__ float sUp[32 * 64];

  const int wv  = threadIdx.x >> 5;
  const int ib0 = blockIdx.x * 64;
  const int n0  = blockIdx.y * 64;
  const int e   = blockIdx.z;

  const float*  wgB = wg + (size_t)e * HDIM * I + ib0;
  const float*  wuB = wu + (size_t)e * HDIM * I + ib0;
  const __bf16* a0  = hb + (size_t)n0 * HDIM;

  v8f cg[4] = {}; v8f cu[4] = {};
  for (int k = 0; k < HDIM; k += 32) {
    stage_panel(sGate, wgB + (size_t)k * I, I);
    stage_panel(sUp,   wuB + (size_t)k * I, I);
    wait_async_all();
    __syncthreads();

    const v16bf bg = frag_b_from_lds(sGate, wv);
    const v16bf bu = frag_b_from_lds(sUp, wv);
#pragma unroll
    for (int m = 0; m < 4; ++m) {
      const v16bf a = load_a_frag(a0 + (size_t)(m * 16) * HDIM + k, HDIM);
      cg[m] = WMMA_BF16(a, bg, cg[m]);
      cu[m] = WMMA_BF16(a, bu, cu[m]);
    }
    __syncthreads();                    // protect LDS before next stage
  }

  const int lane = threadIdx.x & 31;
  const int col  = lane & 15;
  const int rb   = (lane >> 4) << 3;
  const int i0   = ib0 + wv * 16;
#pragma unroll
  for (int m = 0; m < 4; ++m) {
#pragma unroll
    for (int r = 0; r < 8; ++r) {
      const float g = cg[m][r], u = cu[m][r];
      const float s = g / (1.0f + __expf(-g));          // silu
      const int row = n0 + m * 16 + rb + r;
      act[((size_t)e * N_TOK + row) * I + i0 + col] = (__bf16)(s * u);
    }
  }
}

// ---------------------------------------------------------------------------
// Fused down projection: loops all 16 routed experts AND the shared expert
// in-register, scaling each expert's WMMA accumulator by comb[n,e] (routed)
// or sigmoid gate (shared), then stores d_out exactly once (no atomics).
// grid = (H/64, N/64), block = 128.
// ---------------------------------------------------------------------------
__global__ __launch_bounds__(128)
void moe_down_fused_kernel(const __bf16* __restrict__ act,   // [E,N,I_MOE] bf16
                           const float*  __restrict__ wd,    // [E,I_MOE,H] f32
                           const __bf16* __restrict__ acts,  // [N,I_SH] bf16
                           const float*  __restrict__ shd,   // [I_SH,H] f32
                           const float*  __restrict__ comb,  // [N,E]
                           const float*  __restrict__ sgate, // [N]
                           float* __restrict__ out) {        // [N,H]
  __shared__ float sB[32 * 64];

  const int wv   = threadIdx.x >> 5;
  const int hb0  = blockIdx.x * 64;
  const int n0   = blockIdx.y * 64;
  const int lane = threadIdx.x & 31;
  const int col  = lane & 15;
  const int rb   = (lane >> 4) << 3;

  v8f tot[4] = {};

  // ---- routed experts ----
  for (int e = 0; e < NEXP; ++e) {
    const __bf16* aE = act + ((size_t)e * N_TOK + n0) * I_MOE;
    const float*  wB = wd + (size_t)e * I_MOE * HDIM + hb0;
    v8f c[4] = {};
    for (int k = 0; k < I_MOE; k += 32) {
      stage_panel(sB, wB + (size_t)k * HDIM, HDIM);
      wait_async_all();
      __syncthreads();
      const v16bf b = frag_b_from_lds(sB, wv);
#pragma unroll
      for (int m = 0; m < 4; ++m) {
        const v16bf a = load_a_frag(aE + (size_t)(m * 16) * I_MOE + k, I_MOE);
        c[m] = WMMA_BF16(a, b, c[m]);
      }
      __syncthreads();
    }
#pragma unroll
    for (int m = 0; m < 4; ++m) {
#pragma unroll
      for (int r = 0; r < 8; ++r) {
        const int row = n0 + m * 16 + rb + r;
        tot[m][r] += comb[(size_t)row * NEXP + e] * c[m][r];
      }
    }
  }

  // ---- shared expert (sigmoid-gated) ----
  {
    const __bf16* aS = acts + (size_t)n0 * I_SH;
    const float*  wB = shd + hb0;
    v8f c[4] = {};
    for (int k = 0; k < I_SH; k += 32) {
      stage_panel(sB, wB + (size_t)k * HDIM, HDIM);
      wait_async_all();
      __syncthreads();
      const v16bf b = frag_b_from_lds(sB, wv);
#pragma unroll
      for (int m = 0; m < 4; ++m) {
        const v16bf a = load_a_frag(aS + (size_t)(m * 16) * I_SH + k, I_SH);
        c[m] = WMMA_BF16(a, b, c[m]);
      }
      __syncthreads();
    }
#pragma unroll
    for (int m = 0; m < 4; ++m) {
#pragma unroll
      for (int r = 0; r < 8; ++r) {
        const int row = n0 + m * 16 + rb + r;
        tot[m][r] += sgate[row] * c[m][r];
      }
    }
  }

  // ---- single store of the finished 64x64 tile ----
#pragma unroll
  for (int m = 0; m < 4; ++m) {
#pragma unroll
    for (int r = 0; r < 8; ++r) {
      const int row = n0 + m * 16 + rb + r;
      out[(size_t)row * HDIM + hb0 + wv * 16 + col] = tot[m][r];
    }
  }
}

// ---------------------------------------------------------------------------
extern "C" void kernel_launch(void* const* d_in, const int* in_sizes, int n_in,
                              void* d_out, int out_size, void* d_ws, size_t ws_size,
                              hipStream_t stream) {
  const float* h         = (const float*)d_in[0];   // [B,S,H] -> [N,H]
  const float* router_w  = (const float*)d_in[1];   // [E,H]
  const float* w_gate    = (const float*)d_in[2];   // [E,H,I_MOE]
  const float* w_up      = (const float*)d_in[3];   // [E,H,I_MOE]
  const float* w_down    = (const float*)d_in[4];   // [E,I_MOE,H]
  const float* sh_gate   = (const float*)d_in[5];   // [H,I_SH]
  const float* sh_up     = (const float*)d_in[6];   // [H,I_SH]
  const float* sh_down   = (const float*)d_in[7];   // [I_SH,H]
  const float* sh_gate_w = (const float*)d_in[8];   // [H,1]
  float* out = (float*)d_out;                       // [N,H]

  auto align256 = [](size_t x) { return (x + 255) & ~(size_t)255; };
  char* ws = (char*)d_ws;
  float* comb = (float*)ws;   ws += align256((size_t)N_TOK * NEXP * sizeof(float));
  float* sg   = (float*)ws;   ws += align256((size_t)N_TOK * sizeof(float));
  __bf16* hb   = (__bf16*)ws; ws += align256((size_t)N_TOK * HDIM * sizeof(__bf16));
  __bf16* act  = (__bf16*)ws; ws += align256((size_t)NEXP * N_TOK * I_MOE * sizeof(__bf16));
  __bf16* acts = (__bf16*)ws;

  cvt_bf16_kernel<<<1024, 256, 0, stream>>>(h, hb, (size_t)N_TOK * HDIM);
  router_kernel<<<N_TOK / 8, 256, 0, stream>>>(h, router_w, sh_gate_w, comb, sg);

  moe_gateup_kernel<<<dim3(I_MOE / 64, N_TOK / 64, NEXP), 128, 0, stream>>>(
      hb, w_gate, w_up, act, I_MOE);
  moe_gateup_kernel<<<dim3(I_SH / 64, N_TOK / 64, 1), 128, 0, stream>>>(
      hb, sh_gate, sh_up, acts, I_SH);

  moe_down_fused_kernel<<<dim3(HDIM / 64, N_TOK / 64), 128, 0, stream>>>(
      act, w_down, acts, sh_down, comb, sg, out);
}
